// GClassifier_88064009437953
// MI455X (gfx1250) — matrix operand order repaired and checked by reading, not compiled, not run on hardware
//
#include <hip/hip_runtime.h>
#include <hip/hip_fp16.h>

// GCN classifier for MI455X (gfx1250, wave32).
// Inputs (setup_inputs order):
//   0: x          [100000,128] f32
//   1: edge_index [2,1600000]  int32
//   2: edge_attr  [1600000]    f32
//   3: W1 [128,128] f32   4: b1 [128] f32
//   5: W2 [128,128] f32   6: b2 [128] f32
//   7: Wm [128,10]  f32   8: bm [10]  f32
// Output: [10] f32
//
// Workspace (~103.3 MB):
//   T    : [N,128] f32  raw dense-transform output (scatter gather source)
//   AGG  : [N,128] f32  aggregation buffer (self-loop + bias pre-initialized
//                       by the GEMM epilogue, edges atomically accumulated)
//   DEG/DINV : [N] f32
//   WF1/WF2  : pre-swizzled f16 WMMA B-fragments of W1/W2 (32 KB each)
//   POOL : [128] f32

typedef __attribute__((ext_vector_type(16))) _Float16 v16h;
typedef __attribute__((ext_vector_type(8)))  float    v8f;

#define N_NODES   100000
#define N_EDGES   1600000
#define DHID      128
#define NCLS      10
#define M_TILES   (N_NODES / 16)   // 6250, N_NODES divisible by 16

// Hardware f32 atomic add, no return (STOREcnt path), device scope.
__device__ __forceinline__ void hw_atomic_add_f32(float* p, float v) {
  asm volatile("global_atomic_add_f32 %0, %1, off scope:SCOPE_DEV"
               :: "v"(p), "v"(v) : "memory");
}

// ---------------- degree / normalization ----------------

__global__ void k_init_deg(float* __restrict__ deg) {
  int i = blockIdx.x * blockDim.x + threadIdx.x;
  if (i < N_NODES) deg[i] = 1.0f;  // self-loop weight 1
}

__global__ void k_deg_accum(const int* __restrict__ ei,
                            const float* __restrict__ ea,
                            float* __restrict__ deg) {
  int e = blockIdx.x * blockDim.x + threadIdx.x;
  if (e < N_EDGES) {
    int dst = ei[N_EDGES + e];
    hw_atomic_add_f32(&deg[dst], ea[e]);
  }
}

__global__ void k_dinv(const float* __restrict__ deg, float* __restrict__ dinv) {
  int i = blockIdx.x * blockDim.x + threadIdx.x;
  if (i < N_NODES) {
    float d = deg[i];
    dinv[i] = d > 0.0f ? rsqrtf(d) : 0.0f;
  }
}

// ---------------- W pre-swizzle into WMMA B-fragment layout ----------------
// B fragment (16x16x32 f16, wave32): lane holds column n = lane&15;
// lanes 0-15 hold K={0..7,16..23}, lanes 16-31 hold K={8..15,24..31} (matches A).
// Fragment slot index = (ntile*4 + kstep)*32 + lane.

__global__ void k_prep_w(const float* __restrict__ W, v16h* __restrict__ Wf) {
  int slot = blockIdx.x * blockDim.x + threadIdx.x;  // 1024 slots
  if (slot >= 8 * 4 * 32) return;
  int nt   = slot >> 7;
  int rem  = slot & 127;
  int ks   = rem >> 5;
  int lane = rem & 31;
  int n  = nt * 16 + (lane & 15);
  int kb = ks * 32 + ((lane & 16) ? 8 : 0);
  v16h b;
#pragma unroll
  for (int h = 0; h < 16; ++h)
    b[h] = (_Float16)W[(kb + h + (h >= 8 ? 8 : 0)) * DHID + n];
  Wf[slot] = b;
}

// ---------------- WMMA GEMM with fused aggregation-init epilogue ------------
// T   = (relu?)X @ W                      (raw, for the edge gather)
// AGG = T * dinv[row]^2 + bias            (self-loop term + bias, pre-scatter)
// A fragment (16x32 f16, wave32): lane holds row m = lane&15;
// lanes 0-15: K = {0..7, 16..23}; lanes 16-31: K = {8..15, 24..31} (f16 pairs).
// In-place safe for X==AGG: a wave loads all A fragments of its 16 rows
// before its first epilogue store, and no other wave touches those rows.

template <bool RELU>
__global__ __launch_bounds__(256) void k_gemm(const float* __restrict__ X,
                                              const v16h* __restrict__ Wf,
                                              const float* __restrict__ dinv,
                                              const float* __restrict__ bias,
                                              float* __restrict__ T,
                                              float* __restrict__ AGG,
                                              int nTiles) {
  int wave  = threadIdx.x >> 5;
  int lane  = threadIdx.x & 31;
  int mtile = blockIdx.x * 8 + wave;
  if (mtile >= nTiles) return;  // wave-uniform: EXEC stays all-ones for WMMA

  int row  = mtile * 16 + (lane & 15);
  int koff = (lane & 16) ? 8 : 0;
  const float* xrow = X + (size_t)row * DHID;

  v16h a[4];
#pragma unroll
  for (int ks = 0; ks < 4; ++ks) {
    int kb = ks * 32 + koff;
    float buf[16];
    *(float4*)(buf + 0)  = *(const float4*)(xrow + kb);
    *(float4*)(buf + 4)  = *(const float4*)(xrow + kb + 4);
    *(float4*)(buf + 8)  = *(const float4*)(xrow + kb + 16);
    *(float4*)(buf + 12) = *(const float4*)(xrow + kb + 20);
#pragma unroll
    for (int h = 0; h < 16; ++h) {
      float v = buf[h];
      if (RELU) v = fmaxf(v, 0.0f);
      a[ks][h] = (_Float16)v;
    }
  }

  int m0 = mtile * 16 + ((lane & 16) ? 8 : 0);

  // self-loop coefficients for this wave's output rows
  float dv2[8];
#pragma unroll
  for (int j = 0; j < 8; ++j) {
    float dv = dinv[m0 + j];
    dv2[j] = dv * dv;
  }

  // software-pipelined B fragments: next ntile's loads overlap current WMMAs
  v16h bcur[4], bnext[4];
#pragma unroll
  for (int ks = 0; ks < 4; ++ks) bcur[ks] = Wf[ks * 32 + lane];

#pragma unroll
  for (int nt = 0; nt < 8; ++nt) {
    if (nt < 7) {
#pragma unroll
      for (int ks = 0; ks < 4; ++ks)
        bnext[ks] = Wf[((nt + 1) * 4 + ks) * 32 + lane];
    }
    v8f c = {};
#pragma unroll
    for (int ks = 0; ks < 4; ++ks) {
      c = __builtin_amdgcn_wmma_f32_16x16x32_f16(
          /*neg_a=*/false, a[ks], /*neg_b=*/false, bcur[ks],
          /*c_mod=*/(short)0, c, /*reuse_a=*/false, /*reuse_b=*/false);
    }
    int n0 = nt * 16 + (lane & 15);
    float bb = bias[n0];
#pragma unroll
    for (int j = 0; j < 8; ++j) {
      float val = c[j];
      size_t off = (size_t)(m0 + j) * DHID + n0;
      T[off]   = val;
      AGG[off] = val * dv2[j] + bb;
    }
#pragma unroll
    for (int ks = 0; ks < 4; ++ks) bcur[ks] = bnext[ks];
  }
}

// ---------------- edge scatter: wave per edge, float4 gather + f32 atomics ----

__global__ void k_scatter(const float4* __restrict__ T4,
                          const int* __restrict__ ei,
                          const float* __restrict__ ea,
                          const float* __restrict__ dinv,
                          float* __restrict__ AGG) {
  int tid  = blockIdx.x * blockDim.x + threadIdx.x;
  int e    = tid >> 5;
  int lane = tid & 31;
  if (e >= N_EDGES) return;
  int s = ei[e];
  int t = ei[N_EDGES + e];
  float coef = dinv[s] * ea[e] * dinv[t];
  float4 v = T4[(size_t)s * 32 + lane];
  float* out = AGG + (size_t)t * DHID + lane * 4;
  asm volatile("global_atomic_add_f32 %0, %1, off scope:SCOPE_DEV"
               :: "v"(out), "v"(v.x * coef) : "memory");
  asm volatile("global_atomic_add_f32 %0, %1, off offset:4 scope:SCOPE_DEV"
               :: "v"(out), "v"(v.y * coef) : "memory");
  asm volatile("global_atomic_add_f32 %0, %1, off offset:8 scope:SCOPE_DEV"
               :: "v"(out), "v"(v.z * coef) : "memory");
  asm volatile("global_atomic_add_f32 %0, %1, off offset:12 scope:SCOPE_DEV"
               :: "v"(out), "v"(v.w * coef) : "memory");
}

// ---------------- pooling + head ----------------

__global__ void k_zero_pool(float* __restrict__ pooled) {
  if (threadIdx.x < DHID) pooled[threadIdx.x] = 0.0f;
}

__global__ void k_pool(const float* __restrict__ AGG, float* __restrict__ pooled) {
  __shared__ float s[256];
  int col  = threadIdx.x & 127;
  int rsel = threadIdx.x >> 7;
  float acc = 0.0f;
  for (int n = blockIdx.x * 2 + rsel; n < N_NODES; n += gridDim.x * 2)
    acc += fmaxf(AGG[(size_t)n * DHID + col], 0.0f);
  s[threadIdx.x] = acc;
  __syncthreads();
  if (threadIdx.x < 128)
    hw_atomic_add_f32(&pooled[col], s[threadIdx.x] + s[threadIdx.x + 128]);
}

__global__ void k_final(const float* __restrict__ pooled,
                        const float* __restrict__ Wm,
                        const float* __restrict__ bm,
                        float* __restrict__ out) {
  int c = threadIdx.x;
  if (c < NCLS) {
    float acc = 0.0f;
    for (int d = 0; d < DHID; ++d)
      acc += pooled[d] * Wm[d * NCLS + c];
    out[c] = acc * (1.0f / (float)N_NODES) + bm[c];
  }
}

// ---------------- launch ----------------

extern "C" void kernel_launch(void* const* d_in, const int* in_sizes, int n_in,
                              void* d_out, int out_size, void* d_ws, size_t ws_size,
                              hipStream_t stream) {
  const float* x   = (const float*)d_in[0];
  const int*   ei  = (const int*)d_in[1];
  const float* ea  = (const float*)d_in[2];
  const float* W1  = (const float*)d_in[3];
  const float* b1  = (const float*)d_in[4];
  const float* W2  = (const float*)d_in[5];
  const float* b2  = (const float*)d_in[6];
  const float* Wm  = (const float*)d_in[7];
  const float* bm  = (const float*)d_in[8];
  float* out = (float*)d_out;

  char* ws = (char*)d_ws;
  const size_t SZ_MAT = (size_t)N_NODES * DHID * sizeof(float);  // 51.2 MB
  float* T     = (float*)(ws);
  float* AGG   = (float*)(ws + SZ_MAT);
  float* DEG   = (float*)(ws + 2 * SZ_MAT);
  float* DINV  = (float*)(ws + 2 * SZ_MAT + 409600);
  v16h*  WF1   = (v16h*)(ws + 2 * SZ_MAT + 2 * 409600);
  v16h*  WF2   = (v16h*)(ws + 2 * SZ_MAT + 2 * 409600 + 32768);
  float* POOL  = (float*)(ws + 2 * SZ_MAT + 2 * 409600 + 65536);

  const int B = 256;
  // normalization
  k_init_deg<<<(N_NODES + B - 1) / B, B, 0, stream>>>(DEG);
  k_deg_accum<<<(N_EDGES + B - 1) / B, B, 0, stream>>>(ei, ea, DEG);
  k_dinv<<<(N_NODES + B - 1) / B, B, 0, stream>>>(DEG, DINV);
  // weight pre-swizzle to WMMA fragment layout
  k_prep_w<<<4, B, 0, stream>>>(W1, WF1);
  k_prep_w<<<4, B, 0, stream>>>(W2, WF2);

  const int gemmBlocks = (M_TILES + 7) / 8;          // 8 waves (M-tiles) / block
  const int sctBlocks  = (N_EDGES * 32 + B - 1) / B; // wave per edge

  // layer 1: T1 = x@W1; AGG pre-initialized with self-loop + b1; edge scatter
  k_gemm<false><<<gemmBlocks, B, 0, stream>>>(x, WF1, DINV, b1, T, AGG, M_TILES);
  k_scatter<<<sctBlocks, B, 0, stream>>>((const float4*)T, ei, ea, DINV, AGG);
  // layer 2: ReLU fused into A-fragment load; in-place AGG -> (T, AGG)
  k_gemm<true><<<gemmBlocks, B, 0, stream>>>(AGG, WF2, DINV, b2, T, AGG, M_TILES);
  k_scatter<<<sctBlocks, B, 0, stream>>>((const float4*)T, ei, ea, DINV, AGG);
  // mean pool (with ReLU) + linear head
  k_zero_pool<<<1, 128, 0, stream>>>(POOL);
  k_pool<<<512, B, 0, stream>>>(AGG, POOL);
  k_final<<<1, 32, 0, stream>>>(POOL, Wm, bm, out);
}
